// SelfAttention_51531017617772
// MI455X (gfx1250) — matrix operand order repaired and checked by reading, not compiled, not run on hardware
//
#include <hip/hip_runtime.h>
#include <hip/hip_bf16.h>
#include <stdint.h>

// ---------------------------------------------------------------------------
// Self-attention (B=8, T=1024, C=768, H=12, D=64) for gfx1250 (MI455X).
// - All matmuls on v_wmma_f32_16x16x32_f16 (f16 operands, f32 accumulate).
// - GEMM tiles staged into LDS by the Tensor Data Mover (tensor_load_to_lds),
//   double-buffered, synchronized with s_wait_tensorcnt. Manual f16 staging
//   fallback if the TDM builtin is unavailable.
// - Flash attention: one wave per 16-row q tile, KV in blocks of 32.
// ---------------------------------------------------------------------------

typedef __attribute__((ext_vector_type(16))) _Float16 v16h;
typedef __attribute__((ext_vector_type(8)))  _Float16 v8h;
typedef __attribute__((ext_vector_type(8)))  float    v8f;
typedef __attribute__((ext_vector_type(2)))  _Float16 h2;
typedef __attribute__((ext_vector_type(4)))  unsigned int u32x4;
typedef __attribute__((ext_vector_type(8)))  int i32x8;
typedef __attribute__((ext_vector_type(4)))  int i32x4;

#define B_  8
#define T_  1024
#define C_  768
#define H_  12
#define D_  64

#if defined(__has_builtin)
#if __has_builtin(__builtin_amdgcn_tensor_load_to_lds)
#define HAS_TDM 1
#endif
#endif

#ifdef HAS_TDM
// Issue one TDM load of a [64 rows x 32 cols] f16 tile (row stride K elems)
// into LDS at byte offset ldsOff, padded to 40-half (80B) LDS rows.
__device__ __forceinline__ void tdm_load_tile64x32(const _Float16* gtile,
                                                   unsigned ldsOff,
                                                   int K, int rowsTotal)
{
    const unsigned long long ga = (unsigned long long)(uintptr_t)gtile;
    u32x4 g0;
    g0[0] = 1u;                                     // count=1, user mode
    g0[1] = ldsOff;                                 // lds_addr (bytes)
    g0[2] = (unsigned)(ga & 0xffffffffull);         // global_addr[31:0]
    g0[3] = (unsigned)((ga >> 32) & 0x01ffffffull)  // global_addr[56:32]
          | (2u << 30);                             // type = 2 ("image")
    i32x8 g1;
    g1[0] = (1 << 16)    // data_size = 1 (2 bytes)
          | (1 << 20)    // pad_enable
          | (3 << 22)    // pad_interval: 16 DWORDs (one 64B tile row)
          | (3 << 25);   // pad_amount: 4 DWORDs (16B) -> 80B LDS row stride
    g1[1] = (K & 0xffff) << 16;                     // tensor_dim0[15:0]
    g1[2] = ((K >> 16) & 0xffff) | ((rowsTotal & 0xffff) << 16);
    g1[3] = ((rowsTotal >> 16) & 0xffff) | (32 << 16);  // tile_dim0 = 32
    g1[4] = 64;                                     // tile_dim1 = 64, dim2 = 0
    g1[5] = K;                                      // tensor_dim0_stride[31:0]
    g1[6] = 0;
    g1[7] = 0;
    i32x4 g2; g2[0] = 0; g2[1] = 0; g2[2] = 0; g2[3] = 0;
    i32x4 g3 = g2;
#if __clang_major__ >= 23
    i32x8 g4; for (int i = 0; i < 8; ++i) g4[i] = 0;
    __builtin_amdgcn_tensor_load_to_lds(g0, g1, g2, g3, g4, 0);
#else
    __builtin_amdgcn_tensor_load_to_lds(g0, g1, g2, g3, 0);
#endif
}
#endif

// ---------------------------------------------------------------------------
// WMMA GEMM:  Out[M][N] = A[M][K] * Bw[N][K]^T   (f16 in, fp32 out)
// Block = 128 threads (4 waves), tile 64x64, K step 32.
// TDM path: double-buffered LDS tiles filled by tensor_load_to_lds.
// ---------------------------------------------------------------------------
__global__ __launch_bounds__(128) void gemm_f16wmma_kernel(
    const _Float16* __restrict__ A, const _Float16* __restrict__ Bw,
    float* __restrict__ Out, int M, int N, int K)
{
    __shared__ _Float16 As[2][64][40];   // 80B row stride (TDM pad / manual)
    __shared__ _Float16 Bs[2][64][40];

    const int tid  = threadIdx.x;
    const int lane = tid & 31;
    const int wave = tid >> 5;
    const int wm   = (wave >> 1) * 32;
    const int wn   = (wave & 1)  * 32;
    const int ln   = lane & 15;
    const int g    = lane >> 4;
    const int m0   = blockIdx.y * 64;
    const int n0   = blockIdx.x * 64;

    v8f acc[2][2];
    for (int i = 0; i < 2; ++i)
        for (int j = 0; j < 2; ++j)
            for (int r = 0; r < 8; ++r) acc[i][j][r] = 0.0f;

    const int nk = K / 32;

#ifdef HAS_TDM
    const unsigned asOff0 = (unsigned)(uintptr_t)&As[0][0][0];
    const unsigned asOff1 = (unsigned)(uintptr_t)&As[1][0][0];
    const unsigned bsOff0 = (unsigned)(uintptr_t)&Bs[0][0][0];
    const unsigned bsOff1 = (unsigned)(uintptr_t)&Bs[1][0][0];
    if (tid < 32) {   // wave 0 drives the TDM
        tdm_load_tile64x32(&A[(size_t)m0 * K], asOff0, K, M);
        tdm_load_tile64x32(&Bw[(size_t)n0 * K], bsOff0, K, N);
    }
#endif

    for (int kb = 0; kb < nk; ++kb) {
        const int k0  = kb * 32;
        const int cur = kb & 1;

#ifdef HAS_TDM
        if (tid < 32) {
            if (kb + 1 < nk) {
                const int k1 = k0 + 32;
                tdm_load_tile64x32(&A[(size_t)m0 * K + k1],
                                   (kb & 1) ? asOff0 : asOff1, K, M);
                tdm_load_tile64x32(&Bw[(size_t)n0 * K + k1],
                                   (kb & 1) ? bsOff0 : bsOff1, K, N);
                __builtin_amdgcn_s_wait_tensorcnt(2);  // current tile done
            } else {
                __builtin_amdgcn_s_wait_tensorcnt(0);
            }
        }
        __syncthreads();
#else
        // Manual f16 staging: 16B vector copies, 2 iterations per thread.
        for (int i = tid; i < 64 * 4; i += 128) {
            const int row = i >> 2;
            const int c8  = (i & 3) * 8;
            *reinterpret_cast<v8h*>(&As[0][row][c8]) =
                *reinterpret_cast<const v8h*>(&A[(size_t)(m0 + row) * K + k0 + c8]);
            *reinterpret_cast<v8h*>(&Bs[0][row][c8]) =
                *reinterpret_cast<const v8h*>(&Bw[(size_t)(n0 + row) * K + k0 + c8]);
        }
        if (kb + 1 < nk) {   // prefetch next K tile -> global_prefetch_b8
            __builtin_prefetch(&A[(size_t)(m0 + (tid >> 1)) * K + k0 + 32], 0, 1);
            __builtin_prefetch(&Bw[(size_t)(n0 + (tid >> 1)) * K + k0 + 32], 0, 1);
        }
        __syncthreads();
#endif

#ifdef HAS_TDM
        const int buf = cur;
#else
        const int buf = 0;
        (void)cur;
#endif
        v16h af[2], bf[2];
        for (int s = 0; s < 2; ++s) {
            const int mrow = wm + s * 16 + ln;      // A-frag: m = lane%16
            for (int e = 0; e < 16; e += 2) {
                const int kk = e + 16 * g;
                h2 p = *reinterpret_cast<const h2*>(&As[buf][mrow][kk]);
                af[s][e] = p[0]; af[s][e + 1] = p[1];
            }
            const int nrow = wn + s * 16 + ln;      // B-frag: n = lane%16
            const int kbase = 16 * g;               // k = e + 16*(lane/16)
            for (int e = 0; e < 16; e += 2) {
                h2 p = *reinterpret_cast<const h2*>(&Bs[buf][nrow][kbase + e]);
                bf[s][e] = p[0]; bf[s][e + 1] = p[1];
            }
        }

        for (int i = 0; i < 2; ++i)
            for (int j = 0; j < 2; ++j)
                acc[i][j] = __builtin_amdgcn_wmma_f32_16x16x32_f16(
                    false, af[i], false, bf[j], (short)0, acc[i][j], false, false);
        __syncthreads();
    }

    // C/D layout: m = r + 8*(lane/16), n = lane%16
    for (int i = 0; i < 2; ++i)
        for (int j = 0; j < 2; ++j)
            for (int r = 0; r < 8; ++r) {
                const int mm = m0 + wm + i * 16 + r + 8 * g;
                const int nn = n0 + wn + j * 16 + ln;
                Out[(size_t)mm * N + nn] = acc[i][j][r];
            }
}

// ---------------------------------------------------------------------------
// fp32 -> f16 conversion (4 elements per thread)
// ---------------------------------------------------------------------------
__global__ __launch_bounds__(256) void f32_to_f16_kernel(
    const float* __restrict__ in, _Float16* __restrict__ out, int n)
{
    const int i = (blockIdx.x * 256 + threadIdx.x) * 4;
    if (i < n) {
        const float4 v = *reinterpret_cast<const float4*>(&in[i]);
        out[i + 0] = (_Float16)v.x;
        out[i + 1] = (_Float16)v.y;
        out[i + 2] = (_Float16)v.z;
        out[i + 3] = (_Float16)v.w;
    }
}

// ---------------------------------------------------------------------------
// RoPE + head split.  Reads qkv [B*T][3C] fp32, writes:
//   qh, kh : f16 [B,H,T,D] (RoPE applied);  vT : f16 [B,H,D,T]
// ---------------------------------------------------------------------------
__global__ __launch_bounds__(256) void rope_split_kernel(
    const float* __restrict__ qkv,
    _Float16* __restrict__ qh, _Float16* __restrict__ kh,
    _Float16* __restrict__ vT)
{
    const int idx = blockIdx.x * 256 + threadIdx.x;
    const int i   = idx & 31;
    int tmp       = idx >> 5;
    const int h   = tmp % H_;  tmp /= H_;
    const int t   = tmp % T_;
    const int b   = tmp / T_;

    const float invf = __expf(-(float)(2 * i) * (9.210340371976184f / 64.0f));
    const float ang  = (float)t * invf;
    float s, c;
    __sincosf(ang, &s, &c);

    const size_t row  = ((size_t)b * T_ + t) * (3 * C_);
    const int    base = h * D_ + i;

    const float q1 = qkv[row + base],           q2 = qkv[row + base + 32];
    const float k1 = qkv[row + C_ + base],      k2 = qkv[row + C_ + base + 32];
    const float v1 = qkv[row + 2 * C_ + base],  v2 = qkv[row + 2 * C_ + base + 32];

    const size_t head = (size_t)b * H_ + h;
    const size_t qo   = (head * T_ + t) * D_ + i;
    qh[qo]      = (_Float16)(q1 * c - q2 * s);
    qh[qo + 32] = (_Float16)(q2 * c + q1 * s);
    kh[qo]      = (_Float16)(k1 * c - k2 * s);
    kh[qo + 32] = (_Float16)(k2 * c + k1 * s);

    const size_t vo = (head * D_ + i) * T_ + t;
    vT[vo]           = (_Float16)v1;
    vT[vo + 32 * T_] = (_Float16)v2;
}

// ---------------------------------------------------------------------------
// Flash attention: one wave per (b, h, 16-row q tile); KV blocks of 32.
// Writes f16 [B,T,C] so the projection GEMM consumes f16 operands directly.
// ---------------------------------------------------------------------------
__global__ __launch_bounds__(32) void flash_attn_kernel(
    const _Float16* __restrict__ qh, const _Float16* __restrict__ kh,
    const _Float16* __restrict__ vT, _Float16* __restrict__ outBTC)
{
    __shared__ _Float16 p_lds[16][40];

    const int lane = threadIdx.x & 31;
    const int ln   = lane & 15;
    const int g    = lane >> 4;

    const int bid = blockIdx.x;
    const int qt  = bid & 63;
    const int h   = (bid >> 6) % H_;
    const int b   = bid / (64 * H_);
    const int q0  = qt * 16;

    const size_t head = (size_t)b * H_ + h;
    const _Float16* qp = qh + head * (size_t)T_ * D_;
    const _Float16* kp = kh + head * (size_t)T_ * D_;
    const _Float16* vp = vT + head * (size_t)D_ * T_;

    // Q tile as two A-fragments (K chunks 0-31, 32-63); m = lane%16.
    v16h qa[2];
    for (int kc = 0; kc < 2; ++kc)
        for (int e = 0; e < 16; e += 2) {
            const int kk = kc * 32 + e + 16 * g;
            h2 p = *reinterpret_cast<const h2*>(&qp[(size_t)(q0 + ln) * D_ + kk]);
            qa[kc][e] = p[0]; qa[kc][e + 1] = p[1];
        }

    v8f   oacc[4];
    float mrun[8], lrun[8];
    for (int nt = 0; nt < 4; ++nt)
        for (int r = 0; r < 8; ++r) oacc[nt][r] = 0.0f;
    for (int r = 0; r < 8; ++r) { mrun[r] = -1e30f; lrun[r] = 0.0f; }

    const int jmax = (q0 + 15) >> 5;
    for (int jb = 0; jb <= jmax; ++jb) {
        const int kv0 = jb * 32;

        if (jb < jmax) {  // prefetch next KV block -> global_prefetch_b8
            __builtin_prefetch(&kp[(size_t)(kv0 + 32 + lane) * D_], 0, 1);
            __builtin_prefetch(&vp[(size_t)(lane * 2) * T_ + kv0 + 32], 0, 1);
        }

        // S = Q @ K^T : two 16x16 f32 fragments.
        v8f sfr[2];
        for (int nc = 0; nc < 2; ++nc) {
            v8f z;
            for (int r = 0; r < 8; ++r) z[r] = 0.0f;
            const v16h kb0 = *reinterpret_cast<const v16h*>(
                &kp[(size_t)(kv0 + nc * 16 + ln) * D_ + 16 * g]);
            const v16h kb1 = *reinterpret_cast<const v16h*>(
                &kp[(size_t)(kv0 + nc * 16 + ln) * D_ + 32 + 16 * g]);
            z = __builtin_amdgcn_wmma_f32_16x16x32_f16(false, qa[0], false, kb0,
                                                       (short)0, z, false, false);
            z = __builtin_amdgcn_wmma_f32_16x16x32_f16(false, qa[1], false, kb1,
                                                       (short)0, z, false, false);
            sfr[nc] = z;
        }

        // Online softmax; row m = r + 8*g, scale = 1/sqrt(64).
        for (int r = 0; r < 8; ++r) {
            const int rowg = q0 + r + 8 * g;
            float s0 = sfr[0][r] * 0.125f;
            float s1 = sfr[1][r] * 0.125f;
            if (kv0 + ln      > rowg) s0 = -1e30f;
            if (kv0 + 16 + ln > rowg) s1 = -1e30f;

            float mloc = fmaxf(s0, s1);
            for (int off = 1; off < 16; off <<= 1)
                mloc = fmaxf(mloc, __shfl_xor(mloc, off, 16));
            const float mnew  = fmaxf(mrun[r], mloc);
            const float alpha = __expf(mrun[r] - mnew);
            const float p0 = __expf(s0 - mnew);
            const float p1 = __expf(s1 - mnew);
            float rs = p0 + p1;
            for (int off = 1; off < 16; off <<= 1)
                rs += __shfl_xor(rs, off, 16);
            lrun[r] = lrun[r] * alpha + rs;
            mrun[r] = mnew;
            for (int nt = 0; nt < 4; ++nt) oacc[nt][r] *= alpha;

            p_lds[r + 8 * g][ln]      = (_Float16)p0;
            p_lds[r + 8 * g][16 + ln] = (_Float16)p1;
        }
        __syncthreads();

        // P as A-fragment (K = 32 kv rows).
        v16h pa;
        for (int e = 0; e < 16; e += 2) {
            const int kk = e + 16 * g;
            h2 p = *reinterpret_cast<const h2*>(&p_lds[ln][kk]);
            pa[e] = p[0]; pa[e + 1] = p[1];
        }

        // O += P @ V : vT layout -> one contiguous 32B load per B-frag.
        for (int nt = 0; nt < 4; ++nt) {
            const v16h vb = *reinterpret_cast<const v16h*>(
                &vp[(size_t)(nt * 16 + ln) * T_ + kv0 + 16 * g]);
            oacc[nt] = __builtin_amdgcn_wmma_f32_16x16x32_f16(
                false, pa, false, vb, (short)0, oacc[nt], false, false);
        }
        __syncthreads();
    }

    for (int nt = 0; nt < 4; ++nt)
        for (int r = 0; r < 8; ++r) {
            const int row = q0 + r + 8 * g;
            outBTC[((size_t)b * T_ + row) * C_ + h * D_ + nt * 16 + ln] =
                (_Float16)(oacc[nt][r] / lrun[r]);
        }
}

// ---------------------------------------------------------------------------
// Launcher
// ---------------------------------------------------------------------------
extern "C" void kernel_launch(void* const* d_in, const int* in_sizes, int n_in,
                              void* d_out, int out_size, void* d_ws, size_t ws_size,
                              hipStream_t stream)
{
    (void)in_sizes; (void)n_in; (void)out_size; (void)ws_size;

    const float* x      = (const float*)d_in[0];   // [B,T,C]
    const float* w_qkv  = (const float*)d_in[1];   // [3C,C]
    const float* w_proj = (const float*)d_in[2];   // [C,C]
    float*       out    = (float*)d_out;           // [B,T,C]

    char* ws = (char*)d_ws;
    const size_t MT   = (size_t)B_ * T_;                 // 8192 rows
    const size_t QKVB = MT * (3 * C_) * sizeof(float);   // 75,497,472 B
    const size_t HEAD = (size_t)B_ * H_ * T_ * D_;       // 6,291,456 elems

    float*    qkv    = (float*)ws;               // [8192][2304] fp32
    _Float16* attn_h = (_Float16*)ws;            // f16, aliases qkv (dead)
    _Float16* qh     = (_Float16*)(ws + QKVB);
    _Float16* kh     = qh + HEAD;
    _Float16* vT     = kh + HEAD;
    _Float16* x_h    = vT + HEAD;                                // [8192][768]
    _Float16* wq_h   = x_h + MT * C_;                            // [2304][768]
    _Float16* wp_h   = wq_h + (size_t)(3 * C_) * C_;             // [768][768]

    // 0) fp32 -> f16 operand conversion
    {
        const int nx = (int)(MT * C_);
        const int nq = 3 * C_ * C_;
        const int np = C_ * C_;
        f32_to_f16_kernel<<<nx / (256 * 4), 256, 0, stream>>>(x, x_h, nx);
        f32_to_f16_kernel<<<nq / (256 * 4), 256, 0, stream>>>(w_qkv, wq_h, nq);
        f32_to_f16_kernel<<<np / (256 * 4), 256, 0, stream>>>(w_proj, wp_h, np);
    }
    // 1) QKV projection: [8192,768] @ [2304,768]^T -> [8192,2304] fp32
    {
        dim3 grid((3 * C_) / 64, MT / 64);
        gemm_f16wmma_kernel<<<grid, 128, 0, stream>>>(x_h, wq_h, qkv,
                                                      (int)MT, 3 * C_, C_);
    }
    // 2) RoPE + head split + V transpose
    {
        const int total = B_ * T_ * H_ * 32;
        rope_split_kernel<<<total / 256, 256, 0, stream>>>(qkv, qh, kh, vT);
    }
    // 3) Flash attention (writes f16 [B,T,C], aliasing the dead qkv buffer)
    {
        flash_attn_kernel<<<B_ * H_ * (T_ / 16), 32, 0, stream>>>(qh, kh, vT,
                                                                  attn_h);
    }
    // 4) Output projection: [8192,768] @ [768,768]^T -> d_out (fp32)
    {
        dim3 grid(C_ / 64, MT / 64);
        gemm_f16wmma_kernel<<<grid, 128, 0, stream>>>(attn_h, wp_h, out,
                                                      (int)MT, C_, C_);
    }
}
